// AutoregressiveLSTM_9534827397435
// MI455X (gfx1250) — matrix operand order repaired and checked by reading, not compiled, not run on hardware
//
#include <hip/hip_runtime.h>

// ---------------------------------------------------------------------------
// CDNA5 (gfx1250, wave32) persistent 2-layer LSTM + autoregressive decoder.
// All recurrent GEMMs use v_wmma_f32_16x16x32_bf16. Weights are pre-packed
// into WMMA B-fragment order (ISA 7.12.2) by pack_frags; Wih0/Whh0/Wih1
// fragments live in LDS (~318KB of the 320KB WGP LDS), Whh1 streams from L2
// with global_prefetch_b8 lead issued one phase ahead.
// ---------------------------------------------------------------------------

typedef __attribute__((ext_vector_type(16))) __bf16 v16bf;
typedef __attribute__((ext_vector_type(8)))  __bf16 v8bf;
typedef __attribute__((ext_vector_type(8)))  float  v8f;

#define TILE_M   32
#define THREADS  256
#define H_DIM    128
#define G4_DIM   512
#define IN_DIM   8
#define FCH      64
#define OUT_DIM  4
#define SEQ_T    512
#define DEC_T    64
#define BATCH    4096

// workspace offsets (bf16 elements)
#define OFF_WIH0F 0                       // 32 ntiles x 1 ktile  (K=8 padded->32)
#define OFF_WHH0F (16384)                 // 32 x 4
#define OFF_WIH1F (16384 + 65536)         // 32 x 4
#define OFF_WHH1F (16384 + 2*65536)       // 32 x 4
#define OFF_W1F   (16384 + 3*65536)       // 4  x 4  (FC_H=64, K=128)

// ---------------------------------------------------------------------------
// Pack a row-major f32 weight W[N][K] (z = in @ W^T) into bf16 WMMA
// B-fragments: frag(nt,kt) is 512 bf16; element (lane,e) = W[nt*16 + lane%16]
//                                                         [kt*32 + 8*(lane/16) + e%8 + 16*(e/8)]
// ---------------------------------------------------------------------------
__global__ void pack_frags(const float* __restrict__ W, __bf16* __restrict__ dst,
                           int Kcols, int KT, int total) {
  int idx = blockIdx.x * blockDim.x + threadIdx.x;
  if (idx >= total) return;
  int e    = idx & 15;
  int lane = (idx >> 4) & 31;
  int frag = idx >> 9;
  int kt = frag % KT, nt = frag / KT;
  int n = nt * 16 + (lane & 15);
  int k = kt * 32 + ((lane >> 4) << 3) + (e & 7) + ((e >> 3) << 4);
  float v = (k < Kcols) ? W[n * Kcols + k] : 0.0f;
  dst[idx] = (__bf16)v;
}

__device__ __forceinline__ float sig_f(float x) { return 1.0f / (1.0f + __expf(-x)); }
__device__ __forceinline__ float tanh_f(float x) {
  float e = __expf(-2.0f * fabsf(x));
  float t = (1.0f - e) / (1.0f + e);
  return x >= 0.0f ? t : -t;
}

// A-fragment (16x32 bf16, M x K) from a row-major bf16 buffer.
// lane: m = lane%16, half = lane/16.
// elements 0..7  = buf[m][kt*32 + 8*half .. +7]
// elements 8..15 = buf[m][kt*32 + 16 + 8*half .. +7]
__device__ __forceinline__ v16bf load_a(const __bf16* buf, int stride, int mt, int kt, int lane) {
  int m = lane & 15, hb = lane >> 4;
  const __bf16* p = buf + (mt * 16 + m) * stride + kt * 32 + 8 * hb;
  union { v16bf v; v8bf h[2]; } u;
  u.h[0] = *(const v8bf*)(p);
  u.h[1] = *(const v8bf*)(p + 16);
  return u.v;
}

// B-fragment: contiguous 32B per lane from fragment-ordered storage.
__device__ __forceinline__ v16bf load_b(const __bf16* frag, int lane) {
  return *(const v16bf*)(frag + lane * 16);
}

#define WMMA_BF16(a, b, c) \
  __builtin_amdgcn_wmma_f32_16x16x32_bf16(false, (a), false, (b), (short)0, (c), false, false)

__global__ __launch_bounds__(THREADS) void lstm_persist(
    const float* __restrict__ x,
    const float* __restrict__ bih0, const float* __restrict__ bhh0,
    const float* __restrict__ bih1, const float* __restrict__ bhh1,
    const float* __restrict__ gamma, const float* __restrict__ beta,
    const float* __restrict__ mean,  const float* __restrict__ var,
    const float* __restrict__ b1,    const float* __restrict__ W2,
    const float* __restrict__ b2,    const float* __restrict__ Wf,
    const float* __restrict__ bfv,
    const __bf16* __restrict__ ws,
    float* __restrict__ out) {
  extern __shared__ char smem_raw[];
  __bf16* whh0 = (__bf16*)smem_raw;              // 65536 elems (128KB)
  __bf16* wih1 = whh0 + 65536;                   // 65536 elems (128KB)
  __bf16* wih0 = wih1 + 65536;                   // 16384 elems (32KB)
  __bf16* h0b  = wih0 + 16384;                   // 32 x 128
  __bf16* h1b  = h0b + TILE_M * H_DIM;           // 32 x 128
  __bf16* xb   = h1b + TILE_M * H_DIM;           // 32 x 32 (K-padded input)
  __bf16* bnb  = xb  + TILE_M * 32;              // 32 x 128
  __bf16* fcb  = bnb + TILE_M * H_DIM;           // 32 x 64
  float*  ob   = (float*)(fcb + TILE_M * FCH);   // 32 x 4

  const int tid  = threadIdx.x;
  const int lane = tid & 31;
  const int wv   = tid >> 5;                     // wave 0..7
  const int r0   = blockIdx.x * TILE_M;

  // ---- stage LDS-resident weight fragments (Whh0, Wih1, Wih0) -------------
  {
    const float4* s0 = (const float4*)(ws + OFF_WHH0F);
    float4* d0 = (float4*)whh0;
    for (int i = tid; i < 65536 * 2 / 16; i += THREADS) d0[i] = s0[i];
    const float4* s1 = (const float4*)(ws + OFF_WIH1F);
    float4* d1 = (float4*)wih1;
    for (int i = tid; i < 65536 * 2 / 16; i += THREADS) d1[i] = s1[i];
    const float4* s2 = (const float4*)(ws + OFF_WIH0F);
    float4* d2 = (float4*)wih0;
    for (int i = tid; i < 16384 * 2 / 16; i += THREADS) d2[i] = s2[i];
  }
  for (int i = tid; i < TILE_M * H_DIM; i += THREADS) { h0b[i] = (__bf16)0.0f; h1b[i] = (__bf16)0.0f; }
  for (int i = tid; i < TILE_M * 32; i += THREADS) xb[i] = (__bf16)0.0f;
  __syncthreads();

  // wave w owns hidden columns [16w, 16w+16) of every gate: ntile = wv + 8*g
  const int jc = wv * 16 + (lane & 15);
  float bL0[4], bL1[4];
#pragma unroll
  for (int g = 0; g < 4; ++g) {
    bL0[g] = bih0[g * H_DIM + jc] + bhh0[g * H_DIM + jc];
    bL1[g] = bih1[g * H_DIM + jc] + bhh1[g * H_DIM + jc];
  }
  v8f c0[2] = {};   // layer-0 cell state, WMMA C-layout, resident in VGPRs
  v8f c1[2] = {};   // layer-1 cell state

#pragma unroll 1
  for (int t = 0; t < SEQ_T + DEC_T; ++t) {
    const bool dec = (t >= SEQ_T);

    // ---- prefetch this step's Whh1 fragments (consumed by the L1 GEMM,
    //      one full GEMM + gate phase later) : global_prefetch_b8, per-lane
    //      addresses cover the whole 1KB fragment -----------------------------
#pragma unroll
    for (int kt = 0; kt < 4; ++kt) {
#pragma unroll
      for (int g = 0; g < 4; ++g) {
        const __bf16* p = ws + OFF_WHH1F + ((wv + 8 * g) * 4 + kt) * 512 + lane * 16;
        __builtin_prefetch(p, 0, 3);
      }
    }

    // ---- stage x_t (encoder). Decoder xin was written last iteration. ----
    if (!dec) {
      int row = tid >> 3, k = tid & 7;
      xb[row * 32 + k] = (__bf16)x[(size_t)(r0 + row) * (SEQ_T * IN_DIM) + t * IN_DIM + k];
    }
    __syncthreads();

    // ================= layer 0:  z0 = x@Wih0^T + h0@Whh0^T =================
    v8f acc[4][2] = {};
    {
      v16bf ax0 = load_a(xb, 32, 0, 0, lane);
      v16bf ax1 = load_a(xb, 32, 1, 0, lane);
#pragma unroll
      for (int g = 0; g < 4; ++g) {
        int nt = wv + 8 * g;
        v16bf b = load_b(wih0 + nt * 512, lane);             // LDS
        acc[g][0] = WMMA_BF16(ax0, b, acc[g][0]);
        acc[g][1] = WMMA_BF16(ax1, b, acc[g][1]);
      }
#pragma unroll
      for (int kt = 0; kt < 4; ++kt) {
        v16bf a0 = load_a(h0b, H_DIM, 0, kt, lane);
        v16bf a1 = load_a(h0b, H_DIM, 1, kt, lane);
#pragma unroll
        for (int g = 0; g < 4; ++g) {
          int nt = wv + 8 * g;
          v16bf b = load_b(whh0 + (nt * 4 + kt) * 512, lane);  // LDS
          acc[g][0] = WMMA_BF16(a0, b, acc[g][0]);
          acc[g][1] = WMMA_BF16(a1, b, acc[g][1]);
        }
      }
    }
    __syncthreads();  // all waves done reading h0b / xb

    // gates layer 0 (registers only; C-layout: row = e + 8*(lane/16))
#pragma unroll
    for (int mt = 0; mt < 2; ++mt) {
#pragma unroll
      for (int e = 0; e < 8; ++e) {
        float iv = sig_f(acc[0][mt][e] + bL0[0]);
        float fv = sig_f(acc[1][mt][e] + bL0[1]);
        float gv = tanh_f(acc[2][mt][e] + bL0[2]);
        float ov = sig_f(acc[3][mt][e] + bL0[3]);
        float cc = fv * c0[mt][e] + iv * gv;
        c0[mt][e] = cc;
        float hh = ov * tanh_f(cc);
        int m = mt * 16 + e + 8 * (lane >> 4);
        h0b[m * H_DIM + jc] = (__bf16)hh;
      }
    }
    __syncthreads();  // new h0 visible

    // ================= layer 1:  z1 = h0@Wih1^T + h1@Whh1^T ================
    v8f ac2[4][2] = {};
#pragma unroll
    for (int kt = 0; kt < 4; ++kt) {
      v16bf a0 = load_a(h0b, H_DIM, 0, kt, lane);
      v16bf a1 = load_a(h0b, H_DIM, 1, kt, lane);
#pragma unroll
      for (int g = 0; g < 4; ++g) {
        int nt = wv + 8 * g;
        v16bf b = load_b(wih1 + (nt * 4 + kt) * 512, lane);  // LDS
        ac2[g][0] = WMMA_BF16(a0, b, ac2[g][0]);
        ac2[g][1] = WMMA_BF16(a1, b, ac2[g][1]);
      }
    }
#pragma unroll
    for (int kt = 0; kt < 4; ++kt) {
      v16bf a0 = load_a(h1b, H_DIM, 0, kt, lane);
      v16bf a1 = load_a(h1b, H_DIM, 1, kt, lane);
#pragma unroll
      for (int g = 0; g < 4; ++g) {
        int nt = wv + 8 * g;
        v16bf b = load_b(ws + OFF_WHH1F + (nt * 4 + kt) * 512, lane);  // L2 stream
        ac2[g][0] = WMMA_BF16(a0, b, ac2[g][0]);
        ac2[g][1] = WMMA_BF16(a1, b, ac2[g][1]);
      }
    }
    __syncthreads();  // all waves done reading h1b

#pragma unroll
    for (int mt = 0; mt < 2; ++mt) {
#pragma unroll
      for (int e = 0; e < 8; ++e) {
        float iv = sig_f(ac2[0][mt][e] + bL1[0]);
        float fv = sig_f(ac2[1][mt][e] + bL1[1]);
        float gv = tanh_f(ac2[2][mt][e] + bL1[2]);
        float ov = sig_f(ac2[3][mt][e] + bL1[3]);
        float cc = fv * c1[mt][e] + iv * gv;
        c1[mt][e] = cc;
        float hh = ov * tanh_f(cc);
        int m = mt * 16 + e + 8 * (lane >> 4);
        h1b[m * H_DIM + jc] = (__bf16)hh;
      }
    }
    __syncthreads();  // new h1 visible

    // ======================= decoder head (64 steps) =======================
    if (dec) {
      int d = t - SEQ_T;
      // batchnorm -> bnb (bf16)
      for (int i = tid; i < TILE_M * H_DIM; i += THREADS) {
        int j = i & (H_DIM - 1);
        float hv = (float)h1b[i];
        float hn = (hv - mean[j]) * rsqrtf(var[j] + 1e-5f) * gamma[j] + beta[j];
        bnb[i] = (__bf16)hn;
      }
      __syncthreads();
      // fc1 (128 -> 64) via WMMA: wave -> (mt = wv/4, nt = wv%4)
      {
        int mtf = wv >> 2, ntf = wv & 3;
        v8f af = {};
#pragma unroll
        for (int kt = 0; kt < 4; ++kt) {
          v16bf a = load_a(bnb, H_DIM, mtf, kt, lane);
          v16bf b = load_b(ws + OFF_W1F + (ntf * 4 + kt) * 512, lane);
          af = WMMA_BF16(a, b, af);
        }
        int colf = ntf * 16 + (lane & 15);
        float bb = b1[colf];
#pragma unroll
        for (int e = 0; e < 8; ++e) {
          float z = af[e] + bb;
          z = z > 0.0f ? z : 0.0f;                       // relu
          int m = mtf * 16 + e + 8 * (lane >> 4);
          fcb[m * FCH + colf] = (__bf16)z;
        }
      }
      __syncthreads();
      // fc2 (64 -> 4) + output store
      if (tid < TILE_M * OUT_DIM) {
        int row = tid >> 2, oc = tid & 3;
        float s = b2[oc];
        for (int k = 0; k < FCH; ++k) s += (float)fcb[row * FCH + k] * W2[oc * FCH + k];
        out[(size_t)(r0 + row) * (DEC_T * OUT_DIM) + d * OUT_DIM + oc] = s;
        ob[row * OUT_DIM + oc] = s;
      }
      __syncthreads();
      // feedback: xin = out @ Wf^T + bf  (4 -> 8), back into xb
      {
        int row = tid >> 3, ic = tid & 7;
        float v = bfv[ic];
#pragma unroll
        for (int oc = 0; oc < OUT_DIM; ++oc) v += ob[row * OUT_DIM + oc] * Wf[ic * OUT_DIM + oc];
        xb[row * 32 + ic] = (__bf16)v;
      }
      __syncthreads();
    }
  }
}

extern "C" void kernel_launch(void* const* d_in, const int* in_sizes, int n_in,
                              void* d_out, int out_size, void* d_ws, size_t ws_size,
                              hipStream_t stream) {
  (void)in_sizes; (void)n_in; (void)out_size; (void)ws_size;
  const float* x    = (const float*)d_in[0];
  const float* Wih0 = (const float*)d_in[1];
  const float* Whh0 = (const float*)d_in[2];
  const float* bih0 = (const float*)d_in[3];
  const float* bhh0 = (const float*)d_in[4];
  const float* Wih1 = (const float*)d_in[5];
  const float* Whh1 = (const float*)d_in[6];
  const float* bih1 = (const float*)d_in[7];
  const float* bhh1 = (const float*)d_in[8];
  const float* gam  = (const float*)d_in[9];
  const float* bet  = (const float*)d_in[10];
  const float* mean = (const float*)d_in[11];
  const float* var  = (const float*)d_in[12];
  const float* W1   = (const float*)d_in[13];
  const float* b1   = (const float*)d_in[14];
  const float* W2   = (const float*)d_in[15];
  const float* b2   = (const float*)d_in[16];
  const float* Wf   = (const float*)d_in[17];
  const float* bfv  = (const float*)d_in[18];
  __bf16* ws = (__bf16*)d_ws;   // needs ~432KB of workspace for packed weights

  // pack weights into WMMA B-fragment order (bf16)
  {
    int tot;
    tot = 32 * 1 * 512; pack_frags<<<(tot + 255) / 256, 256, 0, stream>>>(Wih0, ws + OFF_WIH0F,   8, 1, tot);
    tot = 32 * 4 * 512; pack_frags<<<(tot + 255) / 256, 256, 0, stream>>>(Whh0, ws + OFF_WHH0F, 128, 4, tot);
    tot = 32 * 4 * 512; pack_frags<<<(tot + 255) / 256, 256, 0, stream>>>(Wih1, ws + OFF_WIH1F, 128, 4, tot);
    tot = 32 * 4 * 512; pack_frags<<<(tot + 255) / 256, 256, 0, stream>>>(Whh1, ws + OFF_WHH1F, 128, 4, tot);
    tot =  4 * 4 * 512; pack_frags<<<(tot + 255) / 256, 256, 0, stream>>>(W1,   ws + OFF_W1F,   128, 4, tot);
  }

  size_t smem_elems = 65536 + 65536 + 16384 + TILE_M * H_DIM * 2 + TILE_M * 32 +
                      TILE_M * H_DIM + TILE_M * FCH;            // bf16 elements
  size_t smem = smem_elems * 2 + TILE_M * OUT_DIM * 4;          // + float outbuf
  lstm_persist<<<BATCH / TILE_M, THREADS, smem, stream>>>(
      x, bih0, bhh0, bih1, bhh1, gam, bet, mean, var,
      b1, W2, b2, Wf, bfv, ws, (float*)d_out);
}
